// MultiHeadGaussianRegressor_52312701665785
// MI455X (gfx1250) — matrix operand order repaired and checked by reading, not compiled, not run on hardware
//
#include <hip/hip_runtime.h>

// ---------------------------------------------------------------------------
// MultiHeadGaussianRegressor on gfx1250 (MI455X):
//   emb  = x @ Wb + bb            (131072x1024 @ 1024x512)  -> WMMA bf16/f32
//   pooled = head(emb @ Wp + bp)  (fused into GEMM epilogue, L2-hot)
//   src    = head(emb . Ws[sid] + bs[sid])
// Outputs concatenated: emb | pooled_mu | pooled_scale | src_mu | src_scale
// ---------------------------------------------------------------------------

typedef __attribute__((ext_vector_type(16))) __bf16 v16bf;
typedef __attribute__((ext_vector_type(8)))  __bf16 v8bf;
typedef __attribute__((ext_vector_type(8)))  float  v8f;
typedef __attribute__((ext_vector_type(4)))  unsigned short v4us;

#define D_IN   1024
#define D_OUT  512

__device__ __forceinline__ unsigned short bf16_rne(float f) {
    unsigned int u = __float_as_uint(f);
    unsigned int r = u + 0x7FFFu + ((u >> 16) & 1u);
    return (unsigned short)(r >> 16);
}

__device__ __forceinline__ v4us pack4(float4 v) {
    v4us r;
    r[0] = bf16_rne(v.x); r[1] = bf16_rne(v.y);
    r[2] = bf16_rne(v.z); r[3] = bf16_rne(v.w);
    return r;
}

__device__ __forceinline__ float softplus_f(float x) {
    return fmaxf(x, 0.0f) + log1pf(expf(-fabsf(x)));
}

// Per-row 4-way dot (length 512) by one wave32; writes the 4 head scalars.
__device__ __forceinline__ void head_row(const float* __restrict__ emb,
                                         const int*   __restrict__ source_ids,
                                         const float* __restrict__ Wp,
                                         const float* __restrict__ bp,
                                         const float* __restrict__ Ws,
                                         const float* __restrict__ bs,
                                         float* __restrict__ pooled_mu,
                                         float* __restrict__ pooled_scale,
                                         float* __restrict__ src_mu,
                                         float* __restrict__ src_scale,
                                         int row, int lane)
{
    const int s = source_ids[row];
    const float4* e4  = (const float4*)(emb + (size_t)row * D_OUT);
    const float4* wp4 = (const float4*)Wp;                             // [d][2] pairs
    const float4* ws4 = (const float4*)(Ws + (size_t)s * D_OUT * 2);

    float p0 = 0.f, p1 = 0.f, q0 = 0.f, q1 = 0.f;
#pragma unroll
    for (int j = 0; j < D_OUT / 128; ++j) {       // 4 iters, coalesced b128
        int i4 = j * 32 + lane;                   // float4 index: d = 4*i4..4*i4+3
        float4 ev = e4[i4];
        float4 a0 = wp4[i4 * 2], a1 = wp4[i4 * 2 + 1];
        float4 b0 = ws4[i4 * 2], b1 = ws4[i4 * 2 + 1];
        p0 = fmaf(ev.x, a0.x, fmaf(ev.y, a0.z, fmaf(ev.z, a1.x, fmaf(ev.w, a1.z, p0))));
        p1 = fmaf(ev.x, a0.y, fmaf(ev.y, a0.w, fmaf(ev.z, a1.y, fmaf(ev.w, a1.w, p1))));
        q0 = fmaf(ev.x, b0.x, fmaf(ev.y, b0.z, fmaf(ev.z, b1.x, fmaf(ev.w, b1.z, q0))));
        q1 = fmaf(ev.x, b0.y, fmaf(ev.y, b0.w, fmaf(ev.z, b1.y, fmaf(ev.w, b1.w, q1))));
    }
#pragma unroll
    for (int off = 16; off > 0; off >>= 1) {      // wave32 butterfly
        p0 += __shfl_xor(p0, off, 32);
        p1 += __shfl_xor(p1, off, 32);
        q0 += __shfl_xor(q0, off, 32);
        q1 += __shfl_xor(q1, off, 32);
    }
    if (lane == 0) {
        pooled_mu[row]    = p0 + bp[0];
        pooled_scale[row] = softplus_f(p1 + bp[1]) + 0.001f;
        src_mu[row]       = q0 + bs[s * 2 + 0];
        src_scale[row]    = softplus_f(q1 + bs[s * 2 + 1]) + 0.001f;
    }
}

// ---------------------------------------------------------------------------
// GEMM: block tile 64(M) x 256(N), K step 32; 8 waves as 2(M) x 4(N);
// each wave: 2x4 WMMA 16x16 tiles. A/B staged in LDS as bf16 with K contiguous;
// fragments use k = 16*h + s (not the ISA table order), loaded back as ds_load_b128 pairs.
// Last-arriving N-block per row-block computes the heads while emb is L2-hot.
// ---------------------------------------------------------------------------
__launch_bounds__(256, 2)
__global__ void gemm_emb_kernel(const float* __restrict__ x,
                                const float* __restrict__ Wb,
                                const float* __restrict__ bb,
                                float* __restrict__ emb,
                                const int*   __restrict__ source_ids,
                                const float* __restrict__ Wp,
                                const float* __restrict__ bp,
                                const float* __restrict__ Ws,
                                const float* __restrict__ bs,
                                float* __restrict__ pooled_mu,
                                float* __restrict__ pooled_scale,
                                float* __restrict__ src_mu,
                                float* __restrict__ src_scale,
                                unsigned* __restrict__ counters,
                                int fuse_heads)
{
    constexpr int BM = 64, BN = 256, BK = 32;
    constexpr int APITCH = 40;   // ushorts per A row    (32 data + pad) = 80 B
    constexpr int BPITCH = 40;   // ushorts per B column (32 data + pad) = 80 B
    constexpr int KTILES = D_IN / BK;

    __shared__ unsigned short Alds[2][BM * APITCH];  // x rows, row-major
    __shared__ unsigned short Blds[2][BN * BPITCH];  // Wb cols, K-contiguous
    __shared__ int s_last;

    const int tid  = threadIdx.x;
    const int lane = tid & 31;
    const int wave = tid >> 5;
    const int wm   = wave >> 2;      // 0..1 : 32-row slab
    const int wn   = wave & 3;       // 0..3 : 64-col slab
    const int ln   = lane & 15;
    const int h    = lane >> 4;

    const int row0 = blockIdx.y * BM;
    const int col0 = blockIdx.x * BN;

    v8f acc[2][4];
#pragma unroll
    for (int mt = 0; mt < 2; ++mt)
#pragma unroll
        for (int nt = 0; nt < 4; ++nt)
            acc[mt][nt] = (v8f){0.f,0.f,0.f,0.f,0.f,0.f,0.f,0.f};

    auto stage = [&](int buf, int k0) {
        // A tile: 64x32 f32 -> bf16, float4 global loads, b64 LDS stores
#pragma unroll
        for (int it = 0; it < (BM * BK / 4) / 256; ++it) {   // 2 iters
            int idx = it * 256 + tid;
            int m = idx >> 3, k = (idx & 7) * 4;
            float4 v = *(const float4*)(x + (size_t)(row0 + m) * D_IN + k0 + k);
            *(v4us*)&Alds[buf][m * APITCH + k] = pack4(v);
        }
        // B tile: 32x256 f32 -> bf16, float4 global loads, column-major LDS
#pragma unroll
        for (int it = 0; it < (BK * BN / 4) / 256; ++it) {   // 8 iters
            int idx = it * 256 + tid;
            int k = idx >> 6, nn = (idx & 63) * 4;
            float4 v = *(const float4*)(Wb + (size_t)(k0 + k) * D_OUT + col0 + nn);
            Blds[buf][(nn + 0) * BPITCH + k] = bf16_rne(v.x);
            Blds[buf][(nn + 1) * BPITCH + k] = bf16_rne(v.y);
            Blds[buf][(nn + 2) * BPITCH + k] = bf16_rne(v.z);
            Blds[buf][(nn + 3) * BPITCH + k] = bf16_rne(v.w);
        }
    };

    stage(0, 0);
    __syncthreads();

    for (int kt = 0; kt < KTILES; ++kt) {
        const int buf = kt & 1;
        if (kt + 2 < KTILES) {           // warm L2 two tiles ahead
            int kp = (kt + 2) * BK;
            __builtin_prefetch(x  + (size_t)(row0 + (tid >> 3)) * D_IN + kp + (tid & 7) * 4, 0, 0);
            __builtin_prefetch(Wb + (size_t)(kp + (tid >> 6))  * D_OUT + col0 + (tid & 63) * 4, 0, 0);
        }
        if (kt + 1 < KTILES) stage(buf ^ 1, (kt + 1) * BK);

        // A frags: lane=row M(ln); 16 consecutive K starting at h*16
        v16bf afrag[2];
#pragma unroll
        for (int mt = 0; mt < 2; ++mt) {
            int m = wm * 32 + mt * 16 + ln;
            v8bf lo = *(const v8bf*)&Alds[buf][m * APITCH + h * 16];
            v8bf hi = *(const v8bf*)&Alds[buf][m * APITCH + h * 16 + 8];
            afrag[mt] = __builtin_shufflevector(lo, hi,
                0,1,2,3,4,5,6,7,8,9,10,11,12,13,14,15);
        }
        // B frags: lane=col N(ln); 16 consecutive K starting at h*16
        v16bf bfrag[4];
#pragma unroll
        for (int nt = 0; nt < 4; ++nt) {
            int nn = wn * 64 + nt * 16 + ln;
            v8bf lo = *(const v8bf*)&Blds[buf][nn * BPITCH + h * 16];
            v8bf hi = *(const v8bf*)&Blds[buf][nn * BPITCH + h * 16 + 8];
            bfrag[nt] = __builtin_shufflevector(lo, hi,
                0,1,2,3,4,5,6,7,8,9,10,11,12,13,14,15);
        }

#pragma unroll
        for (int mt = 0; mt < 2; ++mt)
#pragma unroll
            for (int nt = 0; nt < 4; ++nt)
                acc[mt][nt] = __builtin_amdgcn_wmma_f32_16x16x32_bf16(
                    false, afrag[mt], false, bfrag[nt],
                    (short)0, acc[mt][nt], false, false);

        __syncthreads();
    }

    // Epilogue: D layout -> lane ln = col, VGPR v = row (v + 8*h) within tile
#pragma unroll
    for (int nt = 0; nt < 4; ++nt) {
        int c = col0 + wn * 64 + nt * 16 + ln;
        float bias = bb[c];
#pragma unroll
        for (int mt = 0; mt < 2; ++mt) {
            int rbase = row0 + wm * 32 + mt * 16 + 8 * h;
#pragma unroll
            for (int v = 0; v < 8; ++v)
                emb[(size_t)(rbase + v) * D_OUT + c] = acc[mt][nt][v] + bias;
        }
    }

    if (!fuse_heads) return;

    // ---- last-arriving N-block for this row-block computes the heads ------
    __threadfence();                        // release emb stores device-wide
    if (tid == 0) {
        unsigned prev = atomicAdd(&counters[blockIdx.y], 1u);
        s_last = (prev + 1 == gridDim.x);
    }
    __syncthreads();
    if (!s_last) return;
    __threadfence();                        // acquire the peer block's stores

    // 64 rows, 8 waves -> 8 rows per wave; emb rows are still L2-resident
#pragma unroll 1
    for (int rr = 0; rr < 8; ++rr)
        head_row(emb, source_ids, Wp, bp, Ws, bs,
                 pooled_mu, pooled_scale, src_mu, src_scale,
                 row0 + wave * 8 + rr, lane);
}

// Fallback (only used if d_ws is too small for the counters)
__launch_bounds__(256, 8)
__global__ void heads_kernel(const float* __restrict__ emb,
                             const int*   __restrict__ source_ids,
                             const float* __restrict__ Wp,
                             const float* __restrict__ bp,
                             const float* __restrict__ Ws,
                             const float* __restrict__ bs,
                             float* __restrict__ pooled_mu,
                             float* __restrict__ pooled_scale,
                             float* __restrict__ src_mu,
                             float* __restrict__ src_scale,
                             int n)
{
    int row = blockIdx.x * 8 + (threadIdx.x >> 5);
    if (row >= n) return;
    head_row(emb, source_ids, Wp, bp, Ws, bs,
             pooled_mu, pooled_scale, src_mu, src_scale, row, threadIdx.x & 31);
}

__global__ void zero_counters(unsigned* __restrict__ c, int n)
{
    int i = blockIdx.x * blockDim.x + threadIdx.x;
    if (i < n) c[i] = 0u;
}

// ---------------------------------------------------------------------------
extern "C" void kernel_launch(void* const* d_in, const int* in_sizes, int n_in,
                              void* d_out, int out_size, void* d_ws, size_t ws_size,
                              hipStream_t stream)
{
    const float* x   = (const float*)d_in[0];
    const int*   sid = (const int*)  d_in[1];
    const float* Wb  = (const float*)d_in[2];
    const float* bb  = (const float*)d_in[3];
    const float* Wp  = (const float*)d_in[4];
    const float* bp  = (const float*)d_in[5];
    const float* Ws  = (const float*)d_in[6];
    const float* bs  = (const float*)d_in[7];

    const int n = in_sizes[1];               // 131072 rows

    float* emb          = (float*)d_out;     // outputs concatenated flat
    float* pooled_mu    = emb + (size_t)n * D_OUT;
    float* pooled_scale = pooled_mu + n;
    float* src_mu       = pooled_scale + n;
    float* src_scale    = src_mu + n;

    const int mblocks = n / 64;
    const int fuse = (ws_size >= (size_t)mblocks * sizeof(unsigned)) ? 1 : 0;
    unsigned* counters = (unsigned*)d_ws;

    if (fuse)
        zero_counters<<<dim3((mblocks + 255) / 256), 256, 0, stream>>>(counters, mblocks);

    dim3 gemm_grid(D_OUT / 256, mblocks);    // (2, 2048) blocks of 256 threads
    gemm_emb_kernel<<<gemm_grid, 256, 0, stream>>>(
        x, Wb, bb, emb, sid, Wp, bp, Ws, bs,
        pooled_mu, pooled_scale, src_mu, src_scale, counters, fuse);

    if (!fuse)
        heads_kernel<<<dim3((n + 7) / 8), 256, 0, stream>>>(
            emb, sid, Wp, bp, Ws, bs,
            pooled_mu, pooled_scale, src_mu, src_scale, n);
}
